// SMEFTNet_24635932409880
// MI455X (gfx1250) — matrix operand order repaired and checked by reading, not compiled, not run on hardware
//
#include <hip/hip_runtime.h>
#include <math.h>

// MI455X / gfx1250, wave32. Pair-MLP on WMMA f16 (f32 accum).

typedef _Float16 v16h __attribute__((ext_vector_type(16)));
typedef float    v8f  __attribute__((ext_vector_type(8)));

#define DRN2  0.16f
#define TWOPI 6.283185307179586f

__device__ __forceinline__ v8f wmma16(v16h a, v16h b, v8f c) {
  // D = A(16xK f16) * B(Kx16 f16) + C(16x16 f32), K=32
  return __builtin_amdgcn_wmma_f32_16x16x32_f16(false, a, false, b, (short)0, c,
                                                false, false);
}

// One conv layer. xin: [128 nodes x xs floats] in LDS, with feature offsets
// oPt/oH/oRe/oIm. Writes xout: [128 x 19] = [agg_pt, agg(16), re2, im2].
template <int HIN>
__device__ void conv_layer(int tid,
                           const float* __restrict__ xin, int xs,
                           int oPt, int oH, int oRe, int oIm,
                           float* __restrict__ xout,
                           float* __restrict__ s_winv, float* __restrict__ s_sw,
                           _Float16* __restrict__ s_tile,
                           const float* __restrict__ w0, const float* __restrict__ b0,
                           const float* __restrict__ w1, const float* __restrict__ b1) {
  constexpr int KIN = 3 * HIN + 2;                 // 5 or 50
  const int lane = tid & 31, wave = tid >> 5;

  // ---- per-node normalization: w_ij = adj_ij * pt_i / denom_i ------------
  for (int i = tid; i < 128; i += 256) {
    float ri = xin[i * xs + oRe], ii = xin[i * xs + oIm];
    float deg = 0.f;
    for (int j = 0; j < 128; ++j) {
      float dr = ri - xin[j * xs + oRe], di = ii - xin[j * xs + oIm];
      deg += (dr * dr + di * di <= DRN2) ? 1.f : 0.f;
    }
    float den = xin[i * xs + oPt] * deg;
    s_winv[i] = (den > 0.f) ? xin[i * xs + oPt] / den : 0.f;
    s_sw[i]   = (den > 0.f) ? 1.f : 0.f;
  }
  __syncthreads();

  const int row  = lane & 15;       // A-matrix row (M)
  const int half = lane >> 4;       // K-half selector
  const int koff = half * 8;        // A-layout K offset
  const int colN = lane & 15;       // B/C column (N)
  const int kB   = half * 16;       // unused; B is built with koff
  const int ibase = wave * 16;      // this wave's i-tile
  const int iA = ibase + row;

  // ---- weight matrices in B layout: elem e @ lane = W[(e<8 ? koff+e : 16+koff+e-8)][colN] --------
  v16h B0a, B0b, B1a, B1b;
#pragma unroll
  for (int e = 0; e < 16; ++e) {
    int k = (e < 8) ? (koff + e) : (16 + koff + e - 8);
    B0a[e] = (_Float16)((k < KIN) ? w0[k * 16 + colN] : 0.f);
    int k2 = 32 + k;
    B0b[e] = (_Float16)((k2 < KIN) ? w0[k2 * 16 + colN] : 0.f);
    B1a[e] = (_Float16)((k < 16) ? w1[k * 17 + colN] : 0.f);
    B1b[e] = (_Float16)((k < 16 && colN == 0) ? w1[k * 17 + 16] : 0.f); // gamma col @ N=0
  }
  const float bias0  = b0[colN];
  const float bias1  = b1[colN];
  const float bias1g = b1[16];

  // ---- per-lane loop invariants ------------------------------------------
  float hiA[8];
#pragma unroll
  for (int e = 0; e < 8; ++e)
    hiA[e] = (HIN == 16) ? xin[iA * xs + oH + koff + e] : xin[iA * xs + oH];
  const float reA = xin[iA * xs + oRe], imA = xin[iA * xs + oIm];
  const float r2A = reA * reA + imA * imA;

  float winvC[8], reC[8], imC[8];                  // C-layout rows i = r + 8*half
#pragma unroll
  for (int r = 0; r < 8; ++r) {
    int i = ibase + r + 8 * half;
    winvC[r] = s_winv[i];
    reC[r] = xin[i * xs + oRe];
    imC[r] = xin[i * xs + oIm];
  }

  v8f acc = {}, accg = {};
  _Float16* tile = s_tile + wave * 256;            // per-wave 16x16 f16 scratch

  for (int j = 0; j < 128; ++j) {
    float rej = xin[j * xs + oRe], imj = xin[j * xs + oIm];
    float r2j = rej * rej + imj * imj;
    float nrm = fmaxf(sqrtf(r2A * r2j), 1e-12f);
    float inv = 1.f / nrm;
    float cosv = (reA * rej + imA * imj) * inv;
    float sinv = (imA * rej - reA * imj) * inv;

    // ---- build A matrices: feat = [h_i | h_j | h_j-h_i | cos | sin] ------
    v16h A1, A2;
    if (HIN == 16) {
      float hj[8];
#pragma unroll
      for (int e = 0; e < 8; ++e) hj[e] = xin[j * xs + oH + koff + e];
#pragma unroll
      for (int e = 0; e < 8; ++e) {
        A1[e]     = (_Float16)hiA[e];              // K = koff+e      (h_i)
        A1[e + 8] = (_Float16)hj[e];               // K = 16+koff+e-8 (h_j)
        A2[e]     = (_Float16)(hj[e] - hiA[e]);    // K = 32+koff+e   (h_j-h_i)
        A2[e + 8] = (_Float16)0.f;                 // K = 48.. pad
      }
      if (half == 0) { A2[8] = (_Float16)cosv; A2[9] = (_Float16)sinv; } // K=48,49
    } else {
      float hjv = xin[j * xs + oH];
#pragma unroll
      for (int e = 0; e < 16; ++e) A1[e] = (_Float16)0.f;
      if (half == 0) {
        A1[0] = (_Float16)hiA[0];
        A1[1] = (_Float16)hjv;
        A1[2] = (_Float16)(hjv - hiA[0]);
        A1[3] = (_Float16)cosv;
        A1[4] = (_Float16)sinv;
      }
    }

    // ---- layer 1: hidden = lrelu(feat @ W0 + b0) -------------------------
    v8f c1 = {};
    c1 = wmma16(A1, B0a, c1);
    if (HIN == 16) c1 = wmma16(A2, B0b, c1);
#pragma unroll
    for (int r = 0; r < 8; ++r) {
      float v = c1[r] + bias0;
      c1[r] = (v > 0.f) ? v : 0.01f * v;
    }

    // ---- C-layout -> A-layout via per-wave LDS tile ----------------------
#pragma unroll
    for (int r = 0; r < 8; ++r)
      tile[(r + 8 * half) * 16 + colN] = (_Float16)c1[r];
    asm volatile("s_wait_dscnt 0" ::: "memory");   // same-wave DS in-order + wait
    v16h A3;
#pragma unroll
    for (int e = 0; e < 8; ++e) A3[e] = tile[row * 16 + koff + e];
#pragma unroll
    for (int e = 8; e < 16; ++e) A3[e] = (_Float16)0.f;  // K>=16 pad

    // ---- layer 2: m = hidden @ W1 + b1  (16 cols + gamma col at N=0) -----
    v8f c2 = {}, cg = {};
    c2 = wmma16(A3, B1a, c2);
    cg = wmma16(A3, B1b, cg);

    // ---- weighted aggregation over j -------------------------------------
#pragma unroll
    for (int r = 0; r < 8; ++r) {
      float dr = reC[r] - rej, di = imC[r] - imj;
      float w = (dr * dr + di * di <= DRN2) ? winvC[r] : 0.f;
      acc[r]  += w * (c2[r] + bias1);
      accg[r] += w * (cg[r] + bias1g);
    }
  }

  // ---- epilogue: agg features, pt, gamma-rotated angles ------------------
#pragma unroll
  for (int r = 0; r < 8; ++r) {
    int i = ibase + r + 8 * half;
    xout[i * 19 + 1 + colN] = acc[r];
  }
  if (colN == 0) {                                  // lanes 0 & 16 hold gamma
#pragma unroll
    for (int r = 0; r < 8; ++r) {
      int i = ibase + r + 8 * half;
      float sw = s_sw[i];
      float are = reC[r] * sw, aim = imC[r] * sw;
      float g = accg[r];
      float cs = cosf(TWOPI * g), sn = sinf(TWOPI * g);
      xout[i * 19 + 0]  = xin[i * xs + oPt] * sw;
      xout[i * 19 + 17] = cs * are - sn * aim;
      xout[i * 19 + 18] = sn * are + cs * aim;
    }
  }
  __syncthreads();
}

__global__ __launch_bounds__(256) void smeft_kernel(
    const float* __restrict__ pt, const float* __restrict__ ang,
    const float* __restrict__ c0w0, const float* __restrict__ c0b0,
    const float* __restrict__ c0w1, const float* __restrict__ c0b1,
    const float* __restrict__ c1w0, const float* __restrict__ c1b0,
    const float* __restrict__ c1w1, const float* __restrict__ c1b1,
    const float* __restrict__ rw0, const float* __restrict__ rb0,
    const float* __restrict__ rw1, const float* __restrict__ rb1,
    const float* __restrict__ rw2, const float* __restrict__ rb2,
    float* __restrict__ out) {
  __shared__ float s_x0[128 * 4];     // [pt, |z|, re, im]
  __shared__ float s_x1[128 * 19];    // conv0 out
  __shared__ float s_x2[128 * 19];    // conv1 out
  __shared__ float s_winv[128], s_sw[128];
  __shared__ _Float16 s_tile[8 * 256];
  __shared__ float s_xg[18], s_h1[32], s_h2[32];

  const int b = blockIdx.x, tid = threadIdx.x;

  for (int n = tid; n < 128; n += 256) {
    float p = pt[b * 128 + n];
    float re = ang[(b * 128 + n) * 2 + 0], im = ang[(b * 128 + n) * 2 + 1];
    s_x0[n * 4 + 0] = p;
    s_x0[n * 4 + 1] = sqrtf(re * re + im * im);
    s_x0[n * 4 + 2] = re;
    s_x0[n * 4 + 3] = im;
  }
  __syncthreads();

  conv_layer<1>(tid, s_x0, 4, 0, 1, 2, 3, s_x1, s_winv, s_sw, s_tile,
                c0w0, c0b0, c0w1, c0b1);
  conv_layer<16>(tid, s_x1, 19, 0, 1, 17, 18, s_x2, s_winv, s_sw, s_tile,
                 c1w0, c1b0, c1w1, c1b1);

  // ---- readout: xg = sum_n wj_n * x[n,1:], MLP 16->32->32->1 + sigmoid ----
  if (tid < 18) {
    float S = 0.f;
    for (int n = 0; n < 128; ++n) S += s_x2[n * 19];
    float invS = (S > 0.f) ? 1.f / S : 0.f;
    float a = 0.f;
    for (int n = 0; n < 128; ++n) a += s_x2[n * 19] * invS * s_x2[n * 19 + 1 + tid];
    s_xg[tid] = a;
  }
  __syncthreads();
  if (tid < 32) {
    float a = rb0[tid];
    for (int k = 0; k < 16; ++k) a += s_xg[k] * rw0[k * 32 + tid];
    s_h1[tid] = (a > 0.f) ? a : 0.01f * a;
  }
  __syncthreads();
  if (tid < 32) {
    float a = rb1[tid];
    for (int k = 0; k < 32; ++k) a += s_h1[k] * rw1[k * 32 + tid];
    s_h2[tid] = (a > 0.f) ? a : 0.01f * a;
  }
  __syncthreads();
  if (tid == 0) {
    float a = rb2[0];
    for (int k = 0; k < 32; ++k) a += s_h2[k] * rw2[k];
    out[b * 3 + 0] = 1.f / (1.f + expf(-a));
    out[b * 3 + 1] = s_xg[16];
    out[b * 3 + 2] = s_xg[17];
  }
}

extern "C" void kernel_launch(void* const* d_in, const int* in_sizes, int n_in,
                              void* d_out, int out_size, void* d_ws, size_t ws_size,
                              hipStream_t stream) {
  (void)in_sizes; (void)n_in; (void)d_ws; (void)ws_size; (void)out_size;
  smeft_kernel<<<128, 256, 0, stream>>>(
      (const float*)d_in[0], (const float*)d_in[1],
      (const float*)d_in[2], (const float*)d_in[3],
      (const float*)d_in[4], (const float*)d_in[5],
      (const float*)d_in[6], (const float*)d_in[7],
      (const float*)d_in[8], (const float*)d_in[9],
      (const float*)d_in[10], (const float*)d_in[11],
      (const float*)d_in[12], (const float*)d_in[13],
      (const float*)d_in[14], (const float*)d_in[15],
      (float*)d_out);
}
